// TitansMemory_6107443495532
// MI455X (gfx1250) — compile-verified, matrix operand order checked by reading, not run on hardware
//
#include <hip/hip_runtime.h>
#include <hip/hip_bf16.h>

// ---------------------------------------------------------------------------
// Titans memory update for MI455X (gfx1250, wave32, WMMA).
// bf16 operands + f32 WMMA accumulation (v_wmma_f32_16x16x32_bf16).
// Tile staging: Tensor Data Mover (tensor_load_to_lds, TENSORcnt) when the
// builtin exists, else per-lane global_load_async_to_lds_b128 (ASYNCcnt).
// LDS double buffering either way. Workspace requirement: ~220 MB.
// ---------------------------------------------------------------------------

#define B_   8
#define L_   2048
#define DM_  1024
#define DK_  1024
#define DV_  1024

#define GRAD_CLIP 1.0f
#define MEM_MAX   100.0f
#define SUR_MAX   100.0f

#define LDS_STRIDE 48   // elements: 64B data + 32B pad per row (TDM pad fields)

typedef __bf16 v16bf __attribute__((ext_vector_type(16)));
typedef float  v8f   __attribute__((ext_vector_type(8)));
typedef __attribute__((address_space(3))) unsigned short* lds3_t;

#define ASYNC_WAIT() asm volatile("s_wait_asynccnt 0x0" ::: "memory")

#if __has_builtin(__builtin_amdgcn_tensor_load_to_lds)
#define HAVE_TDM 1
#else
#define HAVE_TDM 0
#endif

__device__ __forceinline__ v8f zero8() {
    v8f z;
#pragma unroll
    for (int i = 0; i < 8; ++i) z[i] = 0.0f;
    return z;
}

__device__ __forceinline__ v8f bfwmma(v16bf a, v16bf b, v8f c) {
    // 8 args: (neg_a, A, neg_b, B, c_mod, C, reuse_a, reuse_b)
    return __builtin_amdgcn_wmma_f32_16x16x32_bf16(false, a, false, b,
                                                   (short)0, c, false, false);
}

__device__ __forceinline__ v16bf frag_ld(const unsigned short* lds, int row, int lane) {
    return *reinterpret_cast<const v16bf*>(
        lds + (row + (lane & 15)) * LDS_STRIDE + ((lane >> 4) << 4));
}

// ---- tile staging ---------------------------------------------------------

#if HAVE_TDM
typedef unsigned int v4u __attribute__((ext_vector_type(4)));
typedef int          v8i __attribute__((ext_vector_type(8)));
typedef int          v4i __attribute__((ext_vector_type(4)));

// One TDM DMA: [rows x 32] bf16 tile, row-major source with leading dim ld,
// into LDS rows of LDS_STRIDE(48) elements: 64B payload (pad_interval=3
// -> every 16 DWORDs) + 32B pad (pad_amount=7 -> 8 DWORDs).
__device__ __forceinline__ void tdm_load_tile(unsigned short* lds,
                                              const __hip_bfloat16* gsrc,
                                              int ld, int rows) {
    const unsigned lds_addr = (unsigned)(unsigned long long)(lds3_t)lds;
    const unsigned long long ga = (unsigned long long)gsrc;
    const unsigned td0 = (unsigned)ld;          // tensor dim0 (elements)
    const unsigned td1 = 1u << 20;              // tensor dim1 (generous)
    v4u g0;
    g0[0] = 1u;                                              // count=1, user D#
    g0[1] = lds_addr;                                        // lds_addr
    g0[2] = (unsigned)ga;                                    // global_addr lo
    g0[3] = (unsigned)((ga >> 32) & 0x1FFFFFFu) | (2u << 30); // addr hi | type=2
    v8i g1;
    g1[0] = (int)((1u << 16) |        // data_size = 2 bytes
                  (1u << 20) |        // pad_enable
                  (3u << 22) |        // pad_interval: 16 DWORDs (64 B)
                  (7u << 25));        // pad_amount:   8 DWORDs (32 B)
    g1[1] = (int)((td0 & 0xFFFFu) << 16);                    // tensor_dim0 lo16
    g1[2] = (int)((td0 >> 16) | ((td1 & 0xFFFFu) << 16));    // d0 hi | d1 lo
    g1[3] = (int)((td1 >> 16) | (32u << 16));                // d1 hi | tile_dim0=32
    g1[4] = (int)(unsigned)rows;                             // tile_dim1 | tile_dim2=0
    g1[5] = (int)td0;                                        // dim0_stride lo32
    g1[6] = 0;                                               // stride hi | d1stride lo
    g1[7] = 0;
    v4i z4 = {0, 0, 0, 0};                                   // 2D: groups 2/3 unused
    v8i z8 = {0, 0, 0, 0, 0, 0, 0, 0};
    // 6-arg form (this toolchain): (g0, g1, g2, g3, g4, cpol)
    __builtin_amdgcn_tensor_load_to_lds(g0, g1, z4, z4, z8, 0);
}
#endif

// Per-lane async fallback: one b128 moves 16B global -> LDS (ASYNCcnt).
__device__ __forceinline__ void async_cp16(unsigned short* lds_dst,
                                           const __hip_bfloat16* gsrc) {
    asm volatile("global_load_async_to_lds_b128 %0, %1, off"
                 :: "v"((lds3_t)lds_dst), "v"(gsrc) : "memory");
}

__device__ __forceinline__ void issue_tiles(unsigned short* la,
                                            const __hip_bfloat16* Asrc, int lda,
                                            unsigned short* lb,
                                            const __hip_bfloat16* Bsrc, int ldb) {
#if HAVE_TDM
    if ((threadIdx.x >> 5) == 0) {        // wave 0 issues both DMAs
        tdm_load_tile(la, Asrc, lda, 128);
        tdm_load_tile(lb, Bsrc, ldb, 64);
    }
#else
    int t = threadIdx.x;
#pragma unroll
    for (int it = 0; it < 2; ++it) {      // A tile: 512 segments
        int idx = t + it * 256;
        int r = idx >> 2, sgm = idx & 3;
        async_cp16(la + r * LDS_STRIDE + sgm * 8, Asrc + (size_t)r * lda + sgm * 8);
    }
    {                                     // B tile: 256 segments
        int r = t >> 2, sgm = t & 3;
        async_cp16(lb + r * LDS_STRIDE + sgm * 8, Bsrc + (size_t)r * ldb + sgm * 8);
    }
#endif
}

__device__ __forceinline__ void tiles_wait() {
#if HAVE_TDM
    if ((threadIdx.x >> 5) == 0) __builtin_amdgcn_s_wait_tensorcnt(0);
#else
    ASYNC_WAIT();
#endif
    __syncthreads();   // publish LDS tiles to all waves / recycle buffers
}

// ---- MMA wave step --------------------------------------------------------

struct WaveCtx { int wm, wn, lane; };

__device__ __forceinline__ WaveCtx wave_ctx() {
    WaveCtx w;
    int wave = threadIdx.x >> 5;
    w.lane = threadIdx.x & 31;
    w.wm = (wave & 3) * 32;   // 4 waves along M
    w.wn = (wave >> 2) * 32;  // 2 waves along N
    return w;
}

__device__ __forceinline__ void mma_step(const unsigned short* lA,
                                         const unsigned short* lB,
                                         const WaveCtx& w, v8f acc[2][2]) {
    v16bf a0 = frag_ld(lA, w.wm,      w.lane);
    v16bf a1 = frag_ld(lA, w.wm + 16, w.lane);
    v16bf b0 = frag_ld(lB, w.wn,      w.lane);
    v16bf b1 = frag_ld(lB, w.wn + 16, w.lane);
    acc[0][0] = bfwmma(a0, b0, acc[0][0]);
    acc[0][1] = bfwmma(a0, b1, acc[0][1]);
    acc[1][0] = bfwmma(a1, b0, acc[1][0]);
    acc[1][1] = bfwmma(a1, b1, acc[1][1]);
}

// ---------------------------------------------------------------------------
// Unified GEMM: C = A[M,K] * Bt[N,K]^T, block tile 128x64, K step 32,
// TDM/async double-buffered LDS staging.
//   EPI 0: bf16 C                          (Wqe, v, q)
//   EPI 1: bf16 C + bf16 C1 = C^T per-b    (k -> kb and kTb)
//   EPI 2: bf16 C1 = (C - Aux)^T           (errT, batched)
//   EPI 3: f32  C * scale                  (grad, out; batched)
// ---------------------------------------------------------------------------
template <int EPI>
__global__ __launch_bounds__(256) void k_gemm(
    const __hip_bfloat16* __restrict__ A, int lda, long long sA,
    const __hip_bfloat16* __restrict__ Bt, int ldb, long long sB,
    float* __restrict__ Cf, __hip_bfloat16* __restrict__ Cb, int ldc, long long sC,
    __hip_bfloat16* __restrict__ C1, int ldc1, long long sC1,
    const __hip_bfloat16* __restrict__ Aux, long long sAux,
    int Kdim, float scale) {
    __shared__ alignas(32) unsigned short lA[2][128 * LDS_STRIDE];
    __shared__ alignas(32) unsigned short lB[2][64 * LDS_STRIDE];

    const int b = blockIdx.z;
    A += (size_t)b * sA;
    Bt += (size_t)b * sB;
    if (EPI == 3) Cf += (size_t)b * sC;
    if (EPI == 0 || EPI == 1) Cb += (size_t)b * sC;
    if (EPI == 1 || EPI == 2) C1 += (size_t)b * sC1;
    if (EPI == 2) Aux += (size_t)b * sAux;

    const int m0 = blockIdx.x * 128, n0 = blockIdx.y * 64;
    WaveCtx w = wave_ctx();
    v8f acc[2][2] = {{zero8(), zero8()}, {zero8(), zero8()}};

    // prologue: stage first K slab
    issue_tiles(lA[0], A + (size_t)m0 * lda, lda,
                lB[0], Bt + (size_t)n0 * ldb, ldb);
    tiles_wait();

    int cur = 0;
    for (int k0 = 0; k0 < Kdim; k0 += 32) {
        if (k0 + 32 < Kdim) {   // stage next slab into the other buffer
            issue_tiles(lA[cur ^ 1], A + (size_t)m0 * lda + k0 + 32, lda,
                        lB[cur ^ 1], Bt + (size_t)n0 * ldb + k0 + 32, ldb);
            __builtin_prefetch(A + (size_t)(m0 + (threadIdx.x >> 1)) * lda + k0 + 64, 0, 2);
        }
        mma_step(lA[cur], lB[cur], w, acc);
        tiles_wait();
        cur ^= 1;
    }

    const int rb = m0 + w.wm + (w.lane >> 4) * 8, cb = n0 + w.wn + (w.lane & 15);
#pragma unroll
    for (int i = 0; i < 2; ++i)
#pragma unroll
        for (int j = 0; j < 2; ++j)
#pragma unroll
            for (int r = 0; r < 8; ++r) {
                const int row = rb + i * 16 + r, col = cb + j * 16;
                float v = acc[i][j][r];
                if (EPI == 0) {
                    Cb[(size_t)row * ldc + col] = __float2bfloat16(v);
                } else if (EPI == 1) {
                    // kb [B*L, DK] + kTb [B][DK, L]  (L_ == 2048)
                    Cb[(size_t)row * ldc + col] = __float2bfloat16(v);
                    int bb = row >> 11, ll = row & 2047;
                    C1[(size_t)bb * DK_ * L_ + (size_t)col * ldc1 + ll] =
                        __float2bfloat16(v);
                } else if (EPI == 2) {
                    float e = v - __bfloat162float(Aux[(size_t)row * ldc + col]);
                    C1[(size_t)col * ldc1 + row] = __float2bfloat16(e);
                } else {
                    Cf[(size_t)row * ldc + col] = v * scale;
                }
            }
}

// ---------------------------------------------------------------------------
// Elementwise / reduction kernels
// ---------------------------------------------------------------------------

__device__ __forceinline__ float block_reduce_sum(float v) {
    __shared__ float red[256];
    int t = threadIdx.x;
    red[t] = v;
    __syncthreads();
#pragma unroll
    for (int s = 128; s > 0; s >>= 1) {
        if (t < s) red[t] += red[t + s];
        __syncthreads();
    }
    return red[0];
}

// x -> bf16 + column mean over L (no atomics)
__global__ __launch_bounds__(256) void k_cvt_x_mean(
    const float* __restrict__ x, __hip_bfloat16* __restrict__ xb,
    float* __restrict__ xmean) {
    const int b = blockIdx.y;
    const int dm = blockIdx.x * 256 + threadIdx.x;
    const float* xp = x + (size_t)b * L_ * DM_ + dm;
    __hip_bfloat16* xo = xb + (size_t)b * L_ * DM_ + dm;
    float s = 0.0f;
    for (int l = 0; l < L_; ++l) {
        float v = xp[(size_t)l * DM_];
        s += v;
        xo[(size_t)l * DM_] = __float2bfloat16(v);
    }
    xmean[b * DM_ + dm] = s * (1.0f / (float)L_);
}

__global__ __launch_bounds__(256) void k_f32_to_bf16(
    const float* __restrict__ src, __hip_bfloat16* __restrict__ dst, size_t n) {
    for (size_t i = (size_t)blockIdx.x * blockDim.x + threadIdx.x; i < n;
         i += (size_t)gridDim.x * blockDim.x)
        dst[i] = __float2bfloat16(src[i]);
}

// f32 [R,C] -> bf16 [C,R] transpose-convert (32x32 LDS tile), grid.z batches
__global__ __launch_bounds__(256) void k_cvt_T(
    const float* __restrict__ src, __hip_bfloat16* __restrict__ dst, int R, int C) {
    const size_t boff = (size_t)blockIdx.z * R * C;
    __shared__ unsigned short tile[32][33];
    const int r0 = blockIdx.y * 32, c0 = blockIdx.x * 32;
    const int tr = threadIdx.x >> 5, tc = threadIdx.x & 31;
#pragma unroll
    for (int i = 0; i < 4; ++i) {
        int r = r0 + tr + i * 8;
        __hip_bfloat16 h = __float2bfloat16(src[boff + (size_t)r * C + c0 + tc]);
        tile[tc][tr + i * 8] = *reinterpret_cast<unsigned short*>(&h);
    }
    __syncthreads();
#pragma unroll
    for (int i = 0; i < 4; ++i) {
        int c = c0 + tr + i * 8;
        unsigned short u = tile[tr + i * 8][tc];
        dst[boff + (size_t)c * R + r0 + tc] = *reinterpret_cast<__hip_bfloat16*>(&u);
    }
}

__global__ __launch_bounds__(256) void k_gates(
    const float* __restrict__ xmean,
    const float* __restrict__ Wa, const float* __restrict__ ba,
    const float* __restrict__ We, const float* __restrict__ be,
    const float* __restrict__ Wt, const float* __restrict__ bt,
    float* __restrict__ gates) {
    const int b = blockIdx.x;
    __shared__ float red[3][256];
    float sa = 0.f, se = 0.f, st = 0.f;
    for (int m = threadIdx.x; m < DM_; m += 256) {
        float xm = xmean[b * DM_ + m];
        sa += xm * Wa[m];
        se += xm * We[m];
        st += xm * Wt[m];
    }
    red[0][threadIdx.x] = sa; red[1][threadIdx.x] = se; red[2][threadIdx.x] = st;
    __syncthreads();
#pragma unroll
    for (int s = 128; s > 0; s >>= 1) {
        if (threadIdx.x < s) {
            red[0][threadIdx.x] += red[0][threadIdx.x + s];
            red[1][threadIdx.x] += red[1][threadIdx.x + s];
            red[2][threadIdx.x] += red[2][threadIdx.x + s];
        }
        __syncthreads();
    }
    if (threadIdx.x == 0) {
        gates[b * 4 + 0] = 1.0f / (1.0f + __expf(-(red[0][0] + ba[0])));  // alpha
        gates[b * 4 + 1] = 1.0f / (1.0f + __expf(-(red[1][0] + be[0])));  // eta
        gates[b * 4 + 2] = 1.0f / (1.0f + __expf(-(red[2][0] + bt[0])));  // theta
    }
}

__global__ __launch_bounds__(256) void k_norm_sq(
    const float* __restrict__ src, float* __restrict__ norms, int perB) {
    const int b = blockIdx.y;
    const float* p = src + (size_t)b * perB;
    float s = 0.f;
    for (size_t i = (size_t)blockIdx.x * blockDim.x + threadIdx.x; i < (size_t)perB;
         i += (size_t)gridDim.x * blockDim.x) {
        float v = p[i];
        s += v * v;
    }
    s = block_reduce_sum(s);
    if (threadIdx.x == 0) atomicAdd(&norms[b], s);
}

__global__ __launch_bounds__(256) void k_upd1(
    const float* __restrict__ grad, const float* __restrict__ S,
    const float* __restrict__ gates, float* __restrict__ norms,
    float* __restrict__ S_raw, int perB) {
    const int b = blockIdx.y;
    const float gscale = fminf(GRAD_CLIP / (sqrtf(norms[b]) + 1e-8f), 1.0f);
    const float eta = gates[b * 4 + 1], theta = gates[b * 4 + 2];
    const size_t base = (size_t)b * perB;
    float ss = 0.f;
    for (size_t i = (size_t)blockIdx.x * blockDim.x + threadIdx.x; i < (size_t)perB;
         i += (size_t)gridDim.x * blockDim.x) {
        float v = eta * S[base + i] - theta * (grad[base + i] * gscale);
        S_raw[base + i] = v;
        ss += v * v;
    }
    ss = block_reduce_sum(ss);
    if (threadIdx.x == 0) atomicAdd(&norms[8 + b], ss);
}

__global__ __launch_bounds__(256) void k_upd2(
    float* __restrict__ S_io, const float* __restrict__ M,
    const float* __restrict__ gates, float* __restrict__ norms,
    float* __restrict__ M_raw, int perB) {
    const int b = blockIdx.y;
    const float sscale = fminf(SUR_MAX / (sqrtf(norms[8 + b]) + 1e-8f), 1.0f);
    const float alpha = gates[b * 4 + 0];
    const size_t base = (size_t)b * perB;
    float mm = 0.f;
    for (size_t i = (size_t)blockIdx.x * blockDim.x + threadIdx.x; i < (size_t)perB;
         i += (size_t)gridDim.x * blockDim.x) {
        float sv = S_io[base + i] * sscale;
        S_io[base + i] = sv;                       // final S_new (in place)
        float mv = (1.0f - alpha) * M[base + i] + sv;
        M_raw[base + i] = mv;
        mm += mv * mv;
    }
    mm = block_reduce_sum(mm);
    if (threadIdx.x == 0) atomicAdd(&norms[16 + b], mm);
}

// M_new = clip(M_raw) in place + bf16 TRANSPOSED copy [DV,DK] for out-GEMM B
__global__ __launch_bounds__(256) void k_upd3(
    float* __restrict__ M_io, const float* __restrict__ norms,
    __hip_bfloat16* __restrict__ MtB, int perB) {
    const int b = blockIdx.y;
    const float mscale = fminf(MEM_MAX / (sqrtf(norms[16 + b]) + 1e-8f), 1.0f);
    const size_t base = (size_t)b * perB;
    for (size_t i = (size_t)blockIdx.x * blockDim.x + threadIdx.x; i < (size_t)perB;
         i += (size_t)gridDim.x * blockDim.x) {
        float mv = M_io[base + i] * mscale;
        M_io[base + i] = mv;
        size_t kk = i / DV_, vv = i % DV_;  // i = kk*DV + vv
        MtB[base + vv * DK_ + kk] = __float2bfloat16(mv);
    }
}

// ---------------------------------------------------------------------------
// Host launch
// ---------------------------------------------------------------------------
extern "C" void kernel_launch(void* const* d_in, const int* in_sizes, int n_in,
                              void* d_out, int out_size, void* d_ws, size_t ws_size,
                              hipStream_t stream) {
    (void)in_sizes; (void)n_in; (void)out_size; (void)ws_size;

    const float* x   = (const float*)d_in[0];
    const float* M   = (const float*)d_in[1];
    const float* S   = (const float*)d_in[2];
    const float* Wk  = (const float*)d_in[3];
    const float* Wv  = (const float*)d_in[4];
    const float* Wq  = (const float*)d_in[5];
    const float* Wqk = (const float*)d_in[6];
    const float* Wa  = (const float*)d_in[7];
    const float* ba  = (const float*)d_in[8];
    const float* We  = (const float*)d_in[9];
    const float* be  = (const float*)d_in[10];
    const float* Wt  = (const float*)d_in[11];
    const float* bt  = (const float*)d_in[12];

    float* out  = (float*)d_out;                                    // [B,L,DV]
    float* outM = out + (size_t)B_ * L_ * DV_;                      // [B,DK,DV]
    float* outS = outM + (size_t)B_ * DK_ * DV_;                    // [B,DK,DV]

    char* wsp = (char*)d_ws;
    auto carve = [&](size_t bytes) {
        char* p = wsp;
        wsp += (bytes + 255) & ~(size_t)255;
        return p;
    };
    const size_t XSZ = (size_t)B_ * L_ * DM_ * 2;   // 32 MB
    const size_t WSZ = (size_t)DK_ * DM_ * 2;       // 2 MB
    __hip_bfloat16* xb   = (__hip_bfloat16*)carve(XSZ);   // errT aliases later
    __hip_bfloat16* kb   = (__hip_bfloat16*)carve(XSZ);   // [B*L, DK]
    __hip_bfloat16* kTb  = (__hip_bfloat16*)carve(XSZ);   // [B][DK, L]
    __hip_bfloat16* qb   = (__hip_bfloat16*)carve(XSZ);   // [B*L, DK]
    __hip_bfloat16* vb   = (__hip_bfloat16*)carve(XSZ);   // [B*L, DV]
    float*          grad = (float*)carve((size_t)B_ * DK_ * DV_ * 4);
    __hip_bfloat16* Wkb  = (__hip_bfloat16*)carve(WSZ);   // [DK, DM]
    __hip_bfloat16* Wvb  = (__hip_bfloat16*)carve(WSZ);   // [DV, DM]
    __hip_bfloat16* WqTb = (__hip_bfloat16*)carve(WSZ);   // [DM, DK] = Wq^T
    __hip_bfloat16* Wqkb = (__hip_bfloat16*)carve(WSZ);   // [DK, DK]
    __hip_bfloat16* Wqeb = (__hip_bfloat16*)carve(WSZ);   // [DK, DM]
    __hip_bfloat16* Mtb  = (__hip_bfloat16*)carve((size_t)B_ * DK_ * DV_ * 2); // [B][DV,DK]
    float*          xmean = (float*)carve((size_t)B_ * DM_ * 4);
    float*          gates = (float*)carve(256);
    float*          norms = (float*)carve(256);
    __hip_bfloat16* errT = xb;   // xb dead once k/v/q are produced; [B][DV, L]

    (void)hipMemsetAsync(norms, 0, 256, stream);

    const dim3 blk(256);
    const long long SKV = (long long)L_ * DK_;      // per-batch [L,DK]/[L,DV]
    const long long SMM = (long long)DK_ * DV_;     // per-batch state

    // Stage 0: conversions, mean, gates
    k_cvt_x_mean<<<dim3(DM_ / 256, B_), blk, 0, stream>>>(x, xb, xmean);
    k_f32_to_bf16<<<2048, blk, 0, stream>>>(Wk,  Wkb,  (size_t)DK_ * DM_);
    k_f32_to_bf16<<<2048, blk, 0, stream>>>(Wv,  Wvb,  (size_t)DV_ * DM_);
    k_f32_to_bf16<<<2048, blk, 0, stream>>>(Wqk, Wqkb, (size_t)DK_ * DK_);
    k_cvt_T<<<dim3(DM_ / 32, DK_ / 32, 1), blk, 0, stream>>>(Wq, WqTb, DK_, DM_);
    k_cvt_T<<<dim3(DV_ / 32, DK_ / 32, B_), blk, 0, stream>>>(M, Mtb, DK_, DV_);
    k_gates<<<B_, blk, 0, stream>>>(xmean, Wa, ba, We, be, Wt, bt, gates);

    // Stage 1: Wqe = Wqk @ Wq  (folds TNT projection, saves one giant GEMM)
    k_gemm<0><<<dim3(DK_ / 128, DM_ / 64, 1), blk, 0, stream>>>(
        Wqkb, DK_, 0, WqTb, DK_, 0, nullptr, Wqeb, DM_, 0,
        nullptr, 0, 0, nullptr, 0, DK_, 1.0f);

    // Stage 2: projections k (dual layout), v, q
    const dim3 gproj((B_ * L_) / 128, DK_ / 64, 1);
    k_gemm<1><<<gproj, blk, 0, stream>>>(
        xb, DM_, 0, Wkb, DM_, 0, nullptr, kb, DK_, 0,
        kTb, L_, 0, nullptr, 0, DM_, 1.0f);
    k_gemm<0><<<gproj, blk, 0, stream>>>(
        xb, DM_, 0, Wvb, DM_, 0, nullptr, vb, DV_, 0,
        nullptr, 0, 0, nullptr, 0, DM_, 1.0f);
    k_gemm<0><<<gproj, blk, 0, stream>>>(
        xb, DM_, 0, Wqeb, DM_, 0, nullptr, qb, DK_, 0,
        nullptr, 0, 0, nullptr, 0, DM_, 1.0f);

    // Stage 3: errT = (k@M - v)^T ; grad = (2/L) k^T @ err
    k_gemm<2><<<dim3(L_ / 128, DV_ / 64, B_), blk, 0, stream>>>(
        kb, DK_, SKV, Mtb, DK_, SMM, nullptr, nullptr, DV_, 0,
        errT, L_, (long long)DV_ * L_, vb, SKV, DK_, 1.0f);
    k_gemm<3><<<dim3(DK_ / 128, DV_ / 64, B_), blk, 0, stream>>>(
        kTb, L_, (long long)DK_ * L_, errT, L_, (long long)DV_ * L_,
        grad, nullptr, DV_, SMM, nullptr, 0, 0, nullptr, 0,
        L_, 2.0f / (float)L_);

    // Stage 4: norm-clipped Titans state update
    const int perB = DK_ * DV_;
    k_norm_sq<<<dim3(64, B_), blk, 0, stream>>>(grad, norms, perB);
    k_upd1<<<dim3(128, B_), blk, 0, stream>>>(grad, S, gates, norms, outS, perB);
    k_upd2<<<dim3(128, B_), blk, 0, stream>>>(outS, M, gates, norms, outM, perB);
    k_upd3<<<dim3(128, B_), blk, 0, stream>>>(outM, norms, Mtb, perB); // Mtb := M_new^T bf16

    // Stage 5: output = q @ M_new
    k_gemm<3><<<dim3(L_ / 128, DV_ / 64, B_), blk, 0, stream>>>(
        qb, DK_, SKV, Mtb, DK_, SMM, out, nullptr, DV_, (long long)L_ * DV_,
        nullptr, 0, 0, nullptr, 0, DK_, 1.0f);
}
